// CausalSelfAttention_43782896616060
// MI455X (gfx1250) — matrix operand order repaired and checked by reading, not compiled, not run on hardware
//
#include <hip/hip_runtime.h>
#include <hip/hip_bf16.h>
#include <math.h>

typedef __attribute__((ext_vector_type(16))) _Float16 v16h;
typedef __attribute__((ext_vector_type(8)))  _Float16 v8h;
typedef __attribute__((ext_vector_type(4)))  _Float16 v4h;
typedef __attribute__((ext_vector_type(8)))  float    v8f;
typedef __attribute__((ext_vector_type(4)))  unsigned int u32x4;
typedef __attribute__((ext_vector_type(8)))  int      i32x8;
typedef __attribute__((ext_vector_type(4)))  int      i32x4;

#define B_ 2
#define T_ 2048
#define C_ 1024
#define H_ 16
#define D_ 64
#define EPS_ 1e-5f

// K index inside a 16x32 f16 A-fragment for element e (0..15) of lane-half `half`
// (per CDNA5 ISA 16-bit A-matrix 16x32 layout).
__device__ __forceinline__ int a_k_index(int e, int half) {
  return ((e >> 3) << 4) + (half << 3) + (e & 7);
}

// TDM: DMA a 32-row x 64-col f16 tile (row-major, stride 64) from global to LDS.
// D# packing per CDNA5 ISA.
__device__ __forceinline__ void tdm_load_tile_32x64f16(_Float16* ldsDst,
                                                       const _Float16* gSrc) {
  uint32_t lds_addr = (uint32_t)(uintptr_t)ldsDst;  // low 32 bits = LDS offset
  uint64_t gaddr = (uint64_t)(uintptr_t)gSrc;
  u32x4 g0;
  g0[0] = 1u;                                        // count=1, user descriptor
  g0[1] = lds_addr;                                  // lds_addr [63:32]
  g0[2] = (uint32_t)gaddr;                           // global_addr [95:64]
  g0[3] = (uint32_t)((gaddr >> 32) & 0x01FFFFFFu)    // global_addr [120:96]
        | (2u << 30);                                // type=2 ("image")
  i32x8 g1;
  g1[0] = (int)(1u << 16);    // workgroup_mask=0 | data_size=1 (2B) | no flags
  g1[1] = (int)(64u << 16);   // atomic_barrier_addr=0 | tensor_dim0[15:0]=64
  g1[2] = (int)(32u << 16);   // tensor_dim0[31:16]=0 | tensor_dim1[15:0]=32
  g1[3] = (int)(64u << 16);   // tensor_dim1[31:16]=0 | tile_dim0=64
  g1[4] = (int)32u;           // tile_dim1=32 | tile_dim2=0
  g1[5] = (int)64u;           // tensor_dim0_stride[31:0]=64
  g1[6] = 0;                  // stride hi | tensor_dim1_stride lo
  g1[7] = 0;
  i32x4 g2; g2[0] = 1; g2[1] = 0; g2[2] = 0; g2[3] = 0;  // tensor_dim2=1
  i32x4 g3; g3[0] = 0; g3[1] = 0; g3[2] = 0; g3[3] = 0;
#if defined(__clang_major__) && (__clang_major__ >= 23)
  i32x8 gz = 0;
  __builtin_amdgcn_tensor_load_to_lds(g0, g1, g2, g3, gz, 0);
#else
  __builtin_amdgcn_tensor_load_to_lds(g0, g1, g2, g3, 0);
#endif
}

// Tiled GEMM: Out[m,n] = sum_k A[m,k]*W[n,k]. A: MxK f32, W: NxK f32 (both row-major).
// 256 threads -> 128x128 block tile; 8 waves (4M x 2N), each wave 32x64 (2x4 WMMA tiles).
// LDS double-buffered f16 tiles; K-step = 32.
__global__ __launch_bounds__(256) void gemm_nt_tiled(const float* __restrict__ A,
                                                     const float* __restrict__ W,
                                                     float* __restrict__ Out,
                                                     int M, int N, int K) {
  __shared__ __align__(32) _Float16 sA[2][128 * 32];
  __shared__ __align__(32) _Float16 sB[2][128 * 32];

  const int tid  = threadIdx.x;
  const int lane = tid & 31;
  const int wave = tid >> 5;
  const int wm = wave >> 1;           // 0..3
  const int wn = wave & 1;            // 0..1
  const int half = lane >> 4;
  const int l16  = lane & 15;

  const int nBlocksN = N >> 7;
  const int blockM = (blockIdx.x / nBlocksN) << 7;
  const int blockN = (blockIdx.x % nBlocksN) << 7;

  const int lr = tid >> 3;            // 0..31 (row within 32-row pass)
  const int lc = (tid & 7) << 2;      // 0..28 (float col)

  v8f acc[2][4];
  #pragma unroll
  for (int i = 0; i < 2; ++i)
    #pragma unroll
    for (int t = 0; t < 4; ++t)
      #pragma unroll
      for (int r = 0; r < 8; ++r) acc[i][t][r] = 0.f;

  float4 ra[4], rb[4];

  auto load_tile = [&](int k0) {
    #pragma unroll
    for (int p = 0; p < 4; ++p) {
      ra[p] = *(const float4*)(A + (size_t)(blockM + p * 32 + lr) * K + k0 + lc);
      rb[p] = *(const float4*)(W + (size_t)(blockN + p * 32 + lr) * K + k0 + lc);
    }
  };
  auto store_tile = [&](int buf) {
    #pragma unroll
    for (int p = 0; p < 4; ++p) {
      v4h ha, hb;
      ha[0] = (_Float16)ra[p].x; ha[1] = (_Float16)ra[p].y;
      ha[2] = (_Float16)ra[p].z; ha[3] = (_Float16)ra[p].w;
      hb[0] = (_Float16)rb[p].x; hb[1] = (_Float16)rb[p].y;
      hb[2] = (_Float16)rb[p].z; hb[3] = (_Float16)rb[p].w;
      *(v4h*)&sA[buf][(p * 32 + lr) * 32 + lc] = ha;
      *(v4h*)&sB[buf][(p * 32 + lr) * 32 + lc] = hb;
    }
  };

  const int nK = K >> 5;
  load_tile(0);
  store_tile(0);

  for (int ks = 0; ks < nK; ++ks) {
    const int cur = ks & 1;
    if (ks + 1 < nK) load_tile((ks + 1) << 5);   // issue next global loads early
    __syncthreads();                              // LDS buf[cur] ready

    // B fragments: contiguous 32B per lane (k = half*16 + e)
    v16h bfr[4];
    #pragma unroll
    for (int t = 0; t < 4; ++t) {
      const _Float16* brow = &sB[cur][(wn * 64 + t * 16 + l16) * 32];
      bfr[t] = *(const v16h*)(brow + half * 16);
    }
    #pragma unroll
    for (int i = 0; i < 2; ++i) {
      // A fragment: two contiguous 16B chunks per lane
      const _Float16* arow = &sA[cur][(wm * 32 + i * 16 + l16) * 32];
      v8h c0 = *(const v8h*)(arow + half * 16);
      v8h c1 = *(const v8h*)(arow + half * 16 + 8);
      v16h afr;
      #pragma unroll
      for (int e = 0; e < 8; ++e) { afr[e] = c0[e]; afr[8 + e] = c1[e]; }
      #pragma unroll
      for (int t = 0; t < 4; ++t)
        acc[i][t] = __builtin_amdgcn_wmma_f32_16x16x32_f16(false, afr, false, bfr[t],
                                                           (short)0, acc[i][t], false, false);
    }
    if (ks + 1 < nK) store_tile(cur ^ 1);         // fill other buffer
  }

  #pragma unroll
  for (int i = 0; i < 2; ++i)
    #pragma unroll
    for (int t = 0; t < 4; ++t)
      #pragma unroll
      for (int r = 0; r < 8; ++r) {
        int m = blockM + wm * 32 + i * 16 + r + (half << 3);
        int n = blockN + wn * 64 + t * 16 + l16;
        Out[(size_t)m * N + n] = acc[i][t][r];
      }
}

// qkv[B,T,3C] f32 -> q,k,v [B,H,T,D] f16 with RoPE on q,k.
__global__ __launch_bounds__(256) void rope_split(const float* __restrict__ qkv,
                                                  const float* __restrict__ freqs,
                                                  _Float16* __restrict__ qh,
                                                  _Float16* __restrict__ kh,
                                                  _Float16* __restrict__ vh) {
  int idx = blockIdx.x * blockDim.x + threadIdx.x;   // [b][h][t][d]
  int d = idx & (D_ - 1);
  int t = (idx >> 6) & (T_ - 1);
  int h = (idx >> 17) & (H_ - 1);
  int b = idx >> 21;
  const float* row = qkv + (size_t)(b * T_ + t) * (3 * C_);
  int col = h * D_ + d;
  vh[idx] = (_Float16)row[2 * C_ + col];
  int dd = (d < D_ / 2) ? d : d - D_ / 2;
  float theta = (float)t * freqs[dd];
  float s, c;
  __sincosf(theta, &s, &c);
  float qe = row[h * D_ + 2 * dd], qo = row[h * D_ + 2 * dd + 1];
  float ke = row[C_ + h * D_ + 2 * dd], ko = row[C_ + h * D_ + 2 * dd + 1];
  qh[idx] = (_Float16)((d < D_ / 2) ? (qe * c - qo * s) : (qe * s + qo * c));
  kh[idx] = (_Float16)((d < D_ / 2) ? (ke * c - ko * s) : (ke * s + ko * c));
}

// Flash attention: one wave owns 16 query rows; key blocks of 32.
// K/V tiles DMA'd into per-wave LDS by the Tensor Data Mover, double-buffered
// on TENSORcnt so block i+1's DMA overlaps block i's softmax + WMMA.
__global__ __launch_bounds__(128) void attn_flash(const _Float16* __restrict__ qh,
                                                  const _Float16* __restrict__ kh,
                                                  const _Float16* __restrict__ vh,
                                                  float* __restrict__ y) {
  __shared__ __align__(32) _Float16 kvTiles[4][2][2][32 * 64];  // [wave][buf][K|V]
  __shared__ _Float16 pls[4][16 * 32];
  const int lane = threadIdx.x & 31;
  const int wave = threadIdx.x >> 5;
  const int wid  = blockIdx.x * 4 + wave;
  const int nQB  = T_ / 16;
  const int qb = wid % nQB;
  const int h  = (wid / nQB) % H_;
  const int b  = wid / (nQB * H_);
  const int half = lane >> 4;
  const int l16  = lane & 15;

  const _Float16* qptr  = qh + ((size_t)((b * H_ + h) * T_) + qb * 16) * D_;
  const _Float16* kbase = kh + (size_t)((b * H_ + h) * T_) * D_;
  const _Float16* vbase = vh + (size_t)((b * H_ + h) * T_) * D_;

  v16h aq[2];
  #pragma unroll
  for (int s = 0; s < 2; ++s)
    #pragma unroll
    for (int e = 0; e < 16; ++e)
      aq[s][e] = qptr[(size_t)l16 * D_ + s * 32 + half * 16 + e];

  float rowmax[8], rowsum[8];
  v8f o[4];
  #pragma unroll
  for (int r = 0; r < 8; ++r) { rowmax[r] = -__builtin_inff(); rowsum[r] = 0.f; }
  #pragma unroll
  for (int f = 0; f < 4; ++f)
    #pragma unroll
    for (int r = 0; r < 8; ++r) o[f][r] = 0.f;

  const int qhi = qb * 16 + 15;
  const float scale = 0.125f;   // 1/sqrt(64)

  // prime buffer 0
  tdm_load_tile_32x64f16(&kvTiles[wave][0][0][0], kbase);
  tdm_load_tile_32x64f16(&kvTiles[wave][0][1][0], vbase);

  int ib = 0;
  for (int k0 = 0; k0 <= qhi; k0 += 32, ib ^= 1) {
    const bool hasNext = (k0 + 32 <= qhi);
    if (hasNext) {
      tdm_load_tile_32x64f16(&kvTiles[wave][ib ^ 1][0][0], kbase + (size_t)(k0 + 32) * D_);
      tdm_load_tile_32x64f16(&kvTiles[wave][ib ^ 1][1][0], vbase + (size_t)(k0 + 32) * D_);
      __builtin_amdgcn_s_wait_tensorcnt(2);   // current buffer's 2 DMAs retired
    } else {
      __builtin_amdgcn_s_wait_tensorcnt(0);
    }
    asm volatile("" ::: "memory");
    const _Float16* kT = &kvTiles[wave][ib][0][0];
    const _Float16* vT = &kvTiles[wave][ib][1][0];

    v8f sc[2];
    #pragma unroll
    for (int t = 0; t < 2; ++t) {
      #pragma unroll
      for (int r = 0; r < 8; ++r) sc[t][r] = 0.f;
      #pragma unroll
      for (int s = 0; s < 2; ++s) {
        // B[k][n] = K[key][feat]; contiguous 32B per lane from LDS
        v16h bk = *(const v16h*)(kT + (t * 16 + l16) * 64 + s * 32 + half * 16);
        sc[t] = __builtin_amdgcn_wmma_f32_16x16x32_f16(false, aq[s], false, bk,
                                                       (short)0, sc[t], false, false);
      }
    }
    // causal mask + scale + online softmax
    #pragma unroll
    for (int r = 0; r < 8; ++r) {
      int m = qb * 16 + r + (half << 3);
      #pragma unroll
      for (int t = 0; t < 2; ++t) {
        int key = k0 + t * 16 + l16;
        float v = sc[t][r] * scale;
        sc[t][r] = (key > m) ? -__builtin_inff() : v;
      }
      float v = fmaxf(sc[0][r], sc[1][r]);
      #pragma unroll
      for (int off = 1; off < 16; off <<= 1)
        v = fmaxf(v, __shfl_xor(v, off, 32));
      float newm = fmaxf(rowmax[r], v);
      float corr = __expf(rowmax[r] - newm);
      rowmax[r] = newm;
      float psum = 0.f;
      #pragma unroll
      for (int t = 0; t < 2; ++t) {
        float p = __expf(sc[t][r] - newm);
        sc[t][r] = p;
        psum += p;
      }
      #pragma unroll
      for (int off = 1; off < 16; off <<= 1)
        psum += __shfl_xor(psum, off, 32);
      rowsum[r] = rowsum[r] * corr + psum;
      #pragma unroll
      for (int f = 0; f < 4; ++f) o[f][r] *= corr;
    }
    // transpose P (C-layout) -> A-layout through per-wave LDS
    _Float16* pl = pls[wave];
    #pragma unroll
    for (int r = 0; r < 8; ++r) {
      int m = r + (half << 3);
      #pragma unroll
      for (int t = 0; t < 2; ++t)
        pl[m * 32 + t * 16 + l16] = (_Float16)sc[t][r];
    }
    asm volatile("s_wait_dscnt 0" ::: "memory");
    v16h pa;
    #pragma unroll
    for (int e = 0; e < 16; ++e)
      pa[e] = pl[l16 * 32 + half * 16 + e];
    // O += P @ V
    #pragma unroll
    for (int f = 0; f < 4; ++f) {
      v16h bv;   // B[k][n] = V[k0+k][f*16+n]
      #pragma unroll
      for (int e = 0; e < 16; ++e)
        bv[e] = vT[(half * 16 + e) * 64 + f * 16 + l16];
      o[f] = __builtin_amdgcn_wmma_f32_16x16x32_f16(false, pa, false, bv,
                                                    (short)0, o[f], false, false);
    }
  }
  #pragma unroll
  for (int r = 0; r < 8; ++r) {
    int m = qb * 16 + r + (half << 3);
    float inv = 1.f / rowsum[r];
    #pragma unroll
    for (int f = 0; f < 4; ++f)
      y[(size_t)(b * T_ + m) * C_ + h * D_ + f * 16 + l16] = o[f][r] * inv;
  }
}

__global__ __launch_bounds__(256) void layernorm_k(const float* __restrict__ yin,
                                                   const float* __restrict__ w,
                                                   const float* __restrict__ bias,
                                                   float* __restrict__ out) {
  __shared__ float red[2][256];
  int row = blockIdx.x;
  const float* rp = yin + (size_t)row * C_;
  float lsum = 0.f, lsq = 0.f, vals[4];
  #pragma unroll
  for (int i = 0; i < 4; ++i) {
    float v = rp[threadIdx.x + i * 256];
    vals[i] = v;
    lsum += v; lsq += v * v;
  }
  red[0][threadIdx.x] = lsum; red[1][threadIdx.x] = lsq;
  __syncthreads();
  for (int s = 128; s > 0; s >>= 1) {
    if ((int)threadIdx.x < s) {
      red[0][threadIdx.x] += red[0][threadIdx.x + s];
      red[1][threadIdx.x] += red[1][threadIdx.x + s];
    }
    __syncthreads();
  }
  float mean = red[0][0] * (1.f / C_);
  float var  = red[1][0] * (1.f / C_) - mean * mean;
  float rstd = rsqrtf(var + EPS_);
  #pragma unroll
  for (int i = 0; i < 4; ++i) {
    int c = threadIdx.x + i * 256;
    out[(size_t)row * C_ + c] = (vals[i] - mean) * rstd * w[c] + bias[c];
  }
}

extern "C" void kernel_launch(void* const* d_in, const int* in_sizes, int n_in,
                              void* d_out, int out_size, void* d_ws, size_t ws_size,
                              hipStream_t stream) {
  const float* x      = (const float*)d_in[0];
  const float* rfreqs = (const float*)d_in[1];
  const float* W_attn = (const float*)d_in[2];
  const float* W_proj = (const float*)d_in[3];
  const float* ln_w   = (const float*)d_in[4];
  const float* ln_b   = (const float*)d_in[5];
  float* out = (float*)d_out;

  // Workspace layout (needs 75,497,472 bytes):
  //   [0, 50331648)          qkv f32 (B*T*3C)  -- dead after rope_split
  //   [50331648, +8388608)   q f16 [B,H,T,D]
  //   [.., +8388608)         k f16
  //   [.., +8388608)         v f16
  //   yatt reuses offset 0 (B*T*C f32), y2 at offset 16777216.
  char* ws = (char*)d_ws;
  float*    qkv  = (float*)ws;
  _Float16* qh   = (_Float16*)(ws + 50331648u);
  _Float16* kh   = (_Float16*)(ws + 50331648u + 8388608u);
  _Float16* vh   = (_Float16*)(ws + 50331648u + 2u * 8388608u);
  float*    yatt = (float*)ws;
  float*    y2   = (float*)(ws + (size_t)B_ * T_ * C_ * 4);

  {
    int M = B_ * T_, N = 3 * C_, K = C_;
    int blocks = (M / 128) * (N / 128);              // 768
    gemm_nt_tiled<<<blocks, 256, 0, stream>>>(x, W_attn, qkv, M, N, K);
  }
  {
    int total = B_ * H_ * T_ * D_;                   // 4194304
    rope_split<<<total / 256, 256, 0, stream>>>(qkv, rfreqs, qh, kh, vh);
  }
  {
    int waves = B_ * H_ * (T_ / 16);                 // 4096
    attn_flash<<<waves / 4, 128, 0, stream>>>(qh, kh, vh, yatt);
  }
  {
    int M = B_ * T_, N = C_, K = C_;
    int blocks = (M / 128) * (N / 128);              // 256
    gemm_nt_tiled<<<blocks, 256, 0, stream>>>(yatt, W_proj, y2, M, N, K);
  }
  layernorm_k<<<B_ * T_, 256, 0, stream>>>(y2, ln_w, ln_b, out);
}
